// HybridMoELoRAAttention_858993459669
// MI455X (gfx1250) — matrix operand order repaired and checked by reading, not compiled, not run on hardware
//
#include <hip/hip_runtime.h>
#include <hip/hip_bf16.h>

// ---------------------------------------------------------------------------
// HybridMoELoRAAttention forward, CDNA5 (gfx1250), wave32, bf16 WMMA pipeline.
// GEMMs + attention: LDS double-buffered tiles staged with async-to-LDS
// (ASYNCcnt), WMMA fed from LDS fragments. Grouped top-2 MoE bucket gather.
// ---------------------------------------------------------------------------

#define B_   2
#define S_   2048
#define H_   2048
#define NH_  16
#define DH_  128
#define E_N  6
#define R_   16
#define RP_  32              // LoRA rank padded to one 32-wide K step
#define M_   (B_ * S_)       // 4096 tokens
#define KSTEP 32
#define SCALING_ 8.0f

typedef __bf16 bf16_t;
typedef __attribute__((ext_vector_type(16))) __bf16 v16bf;
typedef __attribute__((ext_vector_type(8)))  float  v8f;
typedef __attribute__((ext_vector_type(4)))  float  f4;
typedef __attribute__((ext_vector_type(4)))  int    i4;

typedef __attribute__((address_space(1))) i4 gi4;   // global 16B vector
typedef __attribute__((address_space(3))) i4 li4;   // LDS 16B vector

#if __has_builtin(__builtin_amdgcn_global_load_async_to_lds_b128) && \
    __has_builtin(__builtin_amdgcn_s_wait_asynccnt)
#define HAVE_ASYNC_LDS 1
#define WAIT_ASYNC(n) __builtin_amdgcn_s_wait_asynccnt(n)
#else
#define HAVE_ASYNC_LDS 0
#define WAIT_ASYNC(n) ((void)0)
#endif

__device__ __forceinline__ bf16_t f2bf(float f) {
  union { float f; unsigned u; } v; v.f = f;
  unsigned r = v.u + 0x7FFFu + ((v.u >> 16) & 1u);   // round-to-nearest-even
  unsigned short h = (unsigned short)(r >> 16);
  return __builtin_bit_cast(__bf16, h);
}

__device__ __forceinline__ void ld16(void* d, const void* s) {
  *reinterpret_cast<f4*>(d) = *reinterpret_cast<const f4*>(s);
}

// 16-byte global -> LDS copy: async (ASYNCcnt) when available, sync fallback.
__device__ __forceinline__ void cp16_g2l(const bf16_t* g, bf16_t* l) {
#if HAVE_ASYNC_LDS
  __builtin_amdgcn_global_load_async_to_lds_b128((gi4*)g, (li4*)l, 0, 0);
#else
  *reinterpret_cast<f4*>(l) = *reinterpret_cast<const f4*>(g);
#endif
}

__device__ __forceinline__ v8f wmma_bf16(v16bf a, v16bf b, v8f c) {
  // (neg_a, A, neg_b, B, c_mod, C, reuse_a, reuse_b)
  return __builtin_amdgcn_wmma_f32_16x16x32_bf16(false, a, false, b, (short)0, c,
                                                 false, false);
}

// ---------------------------------------------------------------------------
// Elementwise conversions
// ---------------------------------------------------------------------------
__global__ void cvt_bf16_kernel(const float* __restrict__ src,
                                bf16_t* __restrict__ dst, size_t n) {
  size_t i = (size_t)blockIdx.x * blockDim.x + threadIdx.x;
  if (i < n) dst[i] = f2bf(src[i]);
}

// src [rows, kin] -> dst [rows, kout] zero-padded (LoRA-B rank padding)
__global__ void cvt_pad_kernel(const float* __restrict__ src,
                               bf16_t* __restrict__ dst,
                               int rows, int kin, int kout) {
  size_t i = (size_t)blockIdx.x * blockDim.x + threadIdx.x;
  size_t n = (size_t)rows * kout;
  if (i >= n) return;
  int r = (int)(i / kout), c = (int)(i % kout);
  dst[i] = (c < kin) ? f2bf(src[(size_t)r * kin + c]) : f2bf(0.0f);
}

// v fp32 [B*S, H] -> vT bf16 [B, NH, DH, S]
__global__ void vtrans_kernel(const float* __restrict__ v,
                              bf16_t* __restrict__ vt) {
  size_t i = (size_t)blockIdx.x * blockDim.x + threadIdx.x;
  size_t n = (size_t)M_ * H_;
  if (i >= n) return;
  size_t tok = i / H_;
  int c = (int)(i % H_);
  int b = (int)(tok / S_), s = (int)(tok % S_);
  int head = c >> 7, d = c & (DH_ - 1);
  vt[((((size_t)b * NH_ + head) * DH_ + d) * S_) + s] = f2bf(v[i]);
}

// ---------------------------------------------------------------------------
// Router: sigmoid scores over E experts, top-2, softmax weights, bucket scatter
// ---------------------------------------------------------------------------
__global__ void route_kernel(const float* __restrict__ x,
                             const float* __restrict__ gw,
                             int* __restrict__ counts,
                             int* __restrict__ btok,
                             float* __restrict__ bw) {
  int t = blockIdx.x * blockDim.x + threadIdx.x;
  if (t >= M_) return;
  const f4* xr = (const f4*)(x + (size_t)t * H_);
  float acc[E_N];
#pragma unroll
  for (int e = 0; e < E_N; ++e) acc[e] = 0.0f;
  for (int i = 0; i < H_ / 4; ++i) {
    f4 xv = xr[i];
#pragma unroll
    for (int e = 0; e < E_N; ++e) {
      f4 gv = ((const f4*)(gw + (size_t)e * H_))[i];
      acc[e] += xv[0] * gv[0] + xv[1] * gv[1] + xv[2] * gv[2] + xv[3] * gv[3];
    }
  }
  float sc[E_N];
#pragma unroll
  for (int e = 0; e < E_N; ++e) sc[e] = 1.0f / (1.0f + __expf(-acc[e]));
  int i1 = 0;
#pragma unroll
  for (int e = 1; e < E_N; ++e) if (sc[e] > sc[i1]) i1 = e;
  int i2 = (i1 == 0) ? 1 : 0;
#pragma unroll
  for (int e = 0; e < E_N; ++e) if (e != i1 && sc[e] > sc[i2]) i2 = e;
  float e2 = __expf(sc[i2] - sc[i1]);
  float w1 = 1.0f / (1.0f + e2), w2 = e2 / (1.0f + e2);
  int p1 = atomicAdd(&counts[i1], 1);
  btok[i1 * M_ + p1] = t; bw[i1 * M_ + p1] = w1;
  int p2 = atomicAdd(&counts[i2], 1);
  btok[i2 * M_ + p2] = t; bw[i2 * M_ + p2] = w2;
}

// ---------------------------------------------------------------------------
// Unified WMMA GEMM + LoRA with async double-buffered LDS staging.
//   out = A[g(m),:] @ W_e^T + lscale * ((A @ LA_e^T) @ LB_e^T)
// Block: 256 threads = 8 waves (4 M-strips x 2 N-halves); tile 64(M) x 128(N).
// Per K-step: A 64x32 + W 128x32 + LoRA-A 16x32 staged into LDS (4x 16B async
// copies per thread, uniform per wave), double buffered, fenced with
// s_wait_asynccnt 4 + barriers; WMMA fragments read from LDS with hoisted
// loads so the 4-WMMA chain issues back-to-back.
// ---------------------------------------------------------------------------
__global__ void gemm_lora_kernel(const bf16_t* __restrict__ A,
                                 const bf16_t* __restrict__ W,
                                 const bf16_t* __restrict__ LA,
                                 const bf16_t* __restrict__ LB,
                                 float* __restrict__ outF,
                                 bf16_t* __restrict__ outH,
                                 const int* __restrict__ btok,
                                 const int* __restrict__ bcnt,
                                 const float* __restrict__ bw,
                                 int M, int N, int K, float lscale) {
  const int e = blockIdx.z;
  const bf16_t* Wbase = W  + (size_t)e * N * K;
  const bf16_t* LAb   = LA + (size_t)e * R_ * K;
  const bf16_t* LBb   = LB + (size_t)e * N * RP_;
  const int cnt = btok ? bcnt[e] : M;
  const int m0 = blockIdx.y * 64;
  if (m0 >= cnt) return;

  const int tid  = threadIdx.x;
  const int lane = tid & 31;
  const int wid  = tid >> 5;
  const int my = wid & 3, nx = wid >> 2;
  const int n0blk = blockIdx.x * 128;
  const int n0 = n0blk + nx * 64;
  const int m = lane & 15, h = lane >> 4;

  __shared__ __align__(16) bf16_t Asl[2][64][KSTEP];    // 8 KB
  __shared__ __align__(16) bf16_t Wsl[2][128][KSTEP];   // 16 KB
  __shared__ __align__(16) bf16_t LAsl[2][16][KSTEP];   // 2 KB
  __shared__ __align__(16) bf16_t tlds[4][16][RP_];     // 4 KB
  __shared__ int   rowtok[64];
  __shared__ float roww[64];

  if (tid < 64) {
    int r = m0 + tid;
    int valid = (r < cnt);
    rowtok[tid] = btok ? (valid ? btok[e * M + r] : 0) : r;
    roww[tid]   = bw   ? (valid ? bw[e * M + r] : 0.0f) : 1.0f;
  }
  __syncthreads();

  // staging assignments (coalesced 16B chunks)
  const int arow = tid >> 2, achk = tid & 3;            // A: 64 rows x 4 chunks
  const int wrow = tid >> 1, wchk = tid & 1;            // W: 128 rows x 2x32B
  const int lrow = (tid >> 2) & 15, lchk = tid & 3;     // LA: 16 rows (4x dup)
  const bf16_t* agp = A + (size_t)rowtok[arow] * K + achk * 8;
  const bf16_t* wgp = Wbase + (size_t)(n0blk + wrow) * K + wchk * 16;
  const bf16_t* lgp = LAb + (size_t)lrow * K + lchk * 8;

  const int nk = K / KSTEP;
  auto stage = [&](int buf, int k0) {
    cp16_g2l(agp + k0,      &Asl[buf][arow][achk * 8]);
    cp16_g2l(wgp + k0,      &Wsl[buf][wrow][wchk * 16]);
    cp16_g2l(wgp + k0 + 8,  &Wsl[buf][wrow][wchk * 16 + 8]);
    cp16_g2l(lgp + k0,      &LAsl[buf][lrow][lchk * 8]);
  };

  v8f acc[4] = {};
  v8f tacc = {};

  stage(0, 0);
  stage(1, KSTEP);
  for (int kk = 0; kk < nk; ++kk) {
    const int p = kk & 1;
    if (kk + 1 < nk) { WAIT_ASYNC(4); } else { WAIT_ASYNC(0); }
    __syncthreads();   // slice p fully resident; previous reads retired

    v16bf af, bf[4];
    ld16(&af,          &Asl[p][my * 16 + m][h * 16]);
    ld16((f4*)&af + 1, &Asl[p][my * 16 + m][h * 16 + 8]);
#pragma unroll
    for (int j = 0; j < 4; ++j) {
      ld16(&bf[j],          &Wsl[p][nx * 64 + j * 16 + m][h * 16]);
      ld16((f4*)&bf[j] + 1, &Wsl[p][nx * 64 + j * 16 + m][h * 16 + 8]);
    }
#pragma unroll
    for (int j = 0; j < 4; ++j)
      acc[j] = wmma_bf16(af, bf[j], acc[j]);
    if (nx == 0) {  // LoRA-A pass: t = x @ LA^T (N = R = 16), frag from LDS
      v16bf lf;
      ld16(&lf,          &LAsl[p][m][h * 16]);
      ld16((f4*)&lf + 1, &LAsl[p][m][h * 16 + 8]);
      tacc = wmma_bf16(af, lf, tacc);
    }
    __syncthreads();   // all waves done reading slice p
    if (kk + 2 < nk) stage(p, (kk + 2) * KSTEP);
  }

  // Stage lscale*t into LDS in A-fragment-friendly [16][32] layout (zero-padded)
  if (nx == 0) {
#pragma unroll
    for (int i = 0; i < 8; ++i) {
      int row = i + h * 8;
      tlds[my][row][m]      = f2bf(lscale * tacc[i]);
      tlds[my][row][16 + m] = f2bf(0.0f);
    }
  }
  __syncthreads();

  v16bf tf;
  ld16(&tf,          &tlds[my][m][h * 16]);
  ld16((f4*)&tf + 1, &tlds[my][m][h * 16 + 8]);
#pragma unroll
  for (int j = 0; j < 4; ++j) {
    const bf16_t* lbrow = LBb + (size_t)(n0 + j * 16 + m) * RP_ + h * 16;
    v16bf bfr;
    ld16(&bfr,          lbrow);
    ld16((f4*)&bfr + 1, lbrow + 8);
    acc[j] = wmma_bf16(tf, bfr, acc[j]);   // single K=32 step (rank padded)
  }

  if (outH) {  // dense q/k: store bf16 in [B, NH, S, DH]
#pragma unroll
    for (int j = 0; j < 4; ++j)
#pragma unroll
      for (int i = 0; i < 8; ++i) {
        int r = m0 + my * 16 + i + h * 8;
        int c = n0 + j * 16 + m;
        int b = r / S_, s = r % S_;
        int head = c >> 7, d = c & (DH_ - 1);
        outH[((((size_t)b * NH_ + head) * S_ + s) << 7) + d] = f2bf(acc[j][i]);
      }
  } else {     // grouped MoE: weighted scatter-accumulate
#pragma unroll
    for (int j = 0; j < 4; ++j)
#pragma unroll
      for (int i = 0; i < 8; ++i) {
        int rl = my * 16 + i + h * 8;
        if (m0 + rl < cnt) {
          int   t2 = rowtok[rl];
          float wg = roww[rl];
          unsafeAtomicAdd(outF + (size_t)t2 * N + (n0 + j * 16 + m),
                          wg * acc[j][i]);
        }
      }
  }
}

// ---------------------------------------------------------------------------
// Flash-style attention with async double-buffered K/V LDS tiles.
// Block = 8 waves = 128 q rows (16 per wave); streams 64-key tiles:
// K-tile 64x128 (16 KB) + V-tile 128x64 (16 KB) staged by all 256 threads
// (8x 16B async copies each), fenced with s_wait_asynccnt 8 + barriers.
// WMMA scores -> online softmax (16-lane shfl) -> probs via per-wave LDS
// round trip (in-order LDS, no block barrier) -> WMMA with V fragments.
// ---------------------------------------------------------------------------
__global__ void attn_kernel(const bf16_t* __restrict__ qh,
                            const bf16_t* __restrict__ kh,
                            const bf16_t* __restrict__ vt,
                            const float* __restrict__ mask,
                            bf16_t* __restrict__ ctx) {
  const int tid  = threadIdx.x;
  const int lane = tid & 31;
  const int w    = tid >> 5;
  const int b  = blockIdx.z;
  const int hh = blockIdx.y;
  const int q0 = blockIdx.x * 128 + w * 16;
  const int m = lane & 15, h = lane >> 4;
  const float scale = 0.08838834764831845f;  // 1/sqrt(128)

  const bf16_t* qb  = qh + (((size_t)b * NH_ + hh) * S_) * DH_;
  const bf16_t* kb  = kh + (((size_t)b * NH_ + hh) * S_) * DH_;
  const bf16_t* vtb = vt + (((size_t)b * NH_ + hh) * DH_) * S_;

  __shared__ __align__(16) bf16_t Ksl[2][64][DH_];     // 32 KB
  __shared__ __align__(16) bf16_t Vsl[2][DH_][64];     // 32 KB
  __shared__ __align__(16) bf16_t plds[8][16][64];     // 16 KB
  __shared__ __align__(16) float  Msl[S_];             //  8 KB

  // stage mask row once (sync stores; first in-loop barrier publishes it)
  {
    const f4* msrc = (const f4*)(mask + (size_t)b * S_);
    f4* mdst = (f4*)Msl;
    for (int i = tid; i < S_ / 4; i += 256) mdst[i] = msrc[i];
  }

  // q fragments (one-time global loads)
  v16bf qf[4];
  const bf16_t* qrow = qb + (size_t)(q0 + m) * DH_;
#pragma unroll
  for (int kk = 0; kk < 4; ++kk) {
    ld16(&qf[kk],          qrow + kk * 32 + h * 16);
    ld16((f4*)&qf[kk] + 1, qrow + kk * 32 + h * 16 + 8);
  }

  // K/V tile staging: 8x 16B chunks per thread per tile pair
  auto stageKV = [&](int buf, int kt) {
#pragma unroll
    for (int q = 0; q < 4; ++q) {
      int cid = tid * 4 + q;                     // 0..1023
      int kr = cid >> 4, kc = (cid & 15) * 8;    // K: 64 rows x 16 chunks
      cp16_g2l(kb + (size_t)(kt + kr) * DH_ + kc, &Ksl[buf][kr][kc]);
      int vr = cid >> 3, vc = (cid & 7) * 8;     // V: 128 rows x 8 chunks
      cp16_g2l(vtb + (size_t)vr * S_ + kt + vc,   &Vsl[buf][vr][vc]);
    }
  };

  v8f cacc[8] = {};
  float mi[8], li[8];
#pragma unroll
  for (int i = 0; i < 8; ++i) { mi[i] = -3.0e38f; li[i] = 0.0f; }

  const int NT = S_ / 64;   // 32 key tiles
  stageKV(0, 0);
  stageKV(1, 64);
  for (int t = 0; t < NT; ++t) {
    const int p = t & 1;
    const int kt = t * 64;
    if (t + 1 < NT) { WAIT_ASYNC(8); } else { WAIT_ASYNC(0); }
    __syncthreads();   // tiles resident; previous reads retired

    // ---- scores: 16 WMMAs from LDS K fragments (hoisted loads) ----
    v8f sc[4] = {};
#pragma unroll
    for (int j = 0; j < 4; ++j) {
      v16bf kf[4];
#pragma unroll
      for (int kk = 0; kk < 4; ++kk) {
        ld16(&kf[kk],          &Ksl[p][j * 16 + m][kk * 32 + h * 16]);
        ld16((f4*)&kf[kk] + 1, &Ksl[p][j * 16 + m][kk * 32 + h * 16 + 8]);
      }
#pragma unroll
      for (int kk = 0; kk < 4; ++kk)
        sc[j] = wmma_bf16(qf[kk], kf[kk], sc[j]);
    }
#pragma unroll
    for (int j = 0; j < 4; ++j) {
      float mk = -10000.0f * (1.0f - Msl[kt + j * 16 + m]);
#pragma unroll
      for (int i = 0; i < 8; ++i) sc[j][i] = sc[j][i] * scale + mk;
    }
    // ---- online softmax: per-row max / sum across the 16 lanes of a half ----
#pragma unroll
    for (int i = 0; i < 8; ++i) {
      float c = fmaxf(fmaxf(sc[0][i], sc[1][i]), fmaxf(sc[2][i], sc[3][i]));
      c = fmaxf(c, __shfl_xor(c, 1, 16));
      c = fmaxf(c, __shfl_xor(c, 2, 16));
      c = fmaxf(c, __shfl_xor(c, 4, 16));
      c = fmaxf(c, __shfl_xor(c, 8, 16));
      float mn   = fmaxf(mi[i], c);
      float corr = __expf(mi[i] - mn);
      mi[i] = mn;
      float ps = 0.0f;
#pragma unroll
      for (int j = 0; j < 4; ++j) {
        float pr = __expf(sc[j][i] - mn);
        sc[j][i] = pr;
        ps += pr;
      }
      ps += __shfl_xor(ps, 1, 16);
      ps += __shfl_xor(ps, 2, 16);
      ps += __shfl_xor(ps, 4, 16);
      ps += __shfl_xor(ps, 8, 16);
      li[i] = li[i] * corr + ps;
#pragma unroll
      for (int d = 0; d < 8; ++d) cacc[d][i] *= corr;
    }
    // ---- probs -> per-wave LDS (in-order LDS, no block barrier needed) ----
#pragma unroll
    for (int j = 0; j < 4; ++j)
#pragma unroll
      for (int i = 0; i < 8; ++i)
        plds[w][i + h * 8][j * 16 + m] = f2bf(sc[j][i]);
    // ---- ctx: 16 WMMAs from LDS V fragments (grouped loads) ----
#pragma unroll
    for (int st = 0; st < 2; ++st) {
      v16bf pf;
      ld16(&pf,          &plds[w][m][st * 32 + h * 16]);
      ld16((f4*)&pf + 1, &plds[w][m][st * 32 + h * 16 + 8]);
#pragma unroll
      for (int dg = 0; dg < 2; ++dg) {
        v16bf vf[4];
#pragma unroll
        for (int d4 = 0; d4 < 4; ++d4) {
          int d = dg * 4 + d4;
          ld16(&vf[d4],          &Vsl[p][d * 16 + m][st * 32 + h * 16]);
          ld16((f4*)&vf[d4] + 1, &Vsl[p][d * 16 + m][st * 32 + h * 16 + 8]);
        }
#pragma unroll
        for (int d4 = 0; d4 < 4; ++d4)
          cacc[dg * 4 + d4] = wmma_bf16(pf, vf[d4], cacc[dg * 4 + d4]);
      }
    }
    __syncthreads();   // all waves done reading tiles p
    if (t + 2 < NT) stageKV(p, (t + 2) * 64);
  }

#pragma unroll
  for (int d = 0; d < 8; ++d)
#pragma unroll
    for (int i = 0; i < 8; ++i) {
      int row = q0 + i + h * 8;
      size_t tok = (size_t)b * S_ + row;
      int col = hh * DH_ + d * 16 + m;
      ctx[tok * H_ + col] = f2bf(cacc[d][i] / li[i]);
    }
}

// ---------------------------------------------------------------------------
// Host-side launch
// ---------------------------------------------------------------------------
static inline void* carve(char*& p, size_t bytes) {
  void* r = p;
  p += (bytes + 255) & ~(size_t)255;
  return r;
}

extern "C" void kernel_launch(void* const* d_in, const int* in_sizes, int n_in,
                              void* d_out, int out_size, void* d_ws, size_t ws_size,
                              hipStream_t stream) {
  const float* x     = (const float*)d_in[0];
  const float* amask = (const float*)d_in[1];
  const float* wq    = (const float*)d_in[2];
  const float* wk    = (const float*)d_in[3];
  const float* qa    = (const float*)d_in[4];
  const float* qb    = (const float*)d_in[5];
  const float* ka    = (const float*)d_in[6];
  const float* kbm   = (const float*)d_in[7];
  const float* gv    = (const float*)d_in[8];
  const float* go    = (const float*)d_in[9];
  const float* vbase = (const float*)d_in[10];
  const float* va    = (const float*)d_in[11];
  const float* vb    = (const float*)d_in[12];
  const float* obase = (const float*)d_in[13];
  const float* oa    = (const float*)d_in[14];
  const float* ob    = (const float*)d_in[15];
  float* out = (float*)d_out;

  char* p = (char*)d_ws;   // ~245 MB total carved from workspace
  bf16_t* xb     = (bf16_t*)carve(p, (size_t)M_ * H_ * 2);
  bf16_t* wqb    = (bf16_t*)carve(p, (size_t)H_ * H_ * 2);
  bf16_t* wkb    = (bf16_t*)carve(p, (size_t)H_ * H_ * 2);
  bf16_t* qab    = (bf16_t*)carve(p, (size_t)R_ * H_ * 2);
  bf16_t* kab    = (bf16_t*)carve(p, (size_t)R_ * H_ * 2);
  bf16_t* qbp    = (bf16_t*)carve(p, (size_t)H_ * RP_ * 2);
  bf16_t* kbp    = (bf16_t*)carve(p, (size_t)H_ * RP_ * 2);
  bf16_t* vbaseb = (bf16_t*)carve(p, (size_t)E_N * H_ * H_ * 2);
  bf16_t* obaseb = (bf16_t*)carve(p, (size_t)E_N * H_ * H_ * 2);
  bf16_t* vab    = (bf16_t*)carve(p, (size_t)E_N * R_ * H_ * 2);
  bf16_t* oab    = (bf16_t*)carve(p, (size_t)E_N * R_ * H_ * 2);
  bf16_t* vbp    = (bf16_t*)carve(p, (size_t)E_N * H_ * RP_ * 2);
  bf16_t* obp    = (bf16_t*)carve(p, (size_t)E_N * H_ * RP_ * 2);
  bf16_t* qhb    = (bf16_t*)carve(p, (size_t)M_ * H_ * 2);
  bf16_t* khb    = (bf16_t*)carve(p, (size_t)M_ * H_ * 2);
  float*  vout   = (float*) carve(p, (size_t)M_ * H_ * 4);
  bf16_t* vtb    = (bf16_t*)carve(p, (size_t)M_ * H_ * 2);
  bf16_t* ctxb   = (bf16_t*)carve(p, (size_t)M_ * H_ * 2);
  int*    counts = (int*)   carve(p, 2 * E_N * 4);
  int*    btokV  = (int*)   carve(p, (size_t)E_N * M_ * 4);
  int*    btokO  = (int*)   carve(p, (size_t)E_N * M_ * 4);
  float*  bwV    = (float*) carve(p, (size_t)E_N * M_ * 4);
  float*  bwO    = (float*) carve(p, (size_t)E_N * M_ * 4);
  int* cntV = counts;
  int* cntO = counts + E_N;

  const int TB = 256;
  auto blk = [](size_t n) { return (unsigned)((n + 255) / 256); };

  // ---- precision conversion ----
  cvt_bf16_kernel<<<blk((size_t)M_ * H_), TB, 0, stream>>>(x, xb, (size_t)M_ * H_);
  cvt_bf16_kernel<<<blk((size_t)H_ * H_), TB, 0, stream>>>(wq, wqb, (size_t)H_ * H_);
  cvt_bf16_kernel<<<blk((size_t)H_ * H_), TB, 0, stream>>>(wk, wkb, (size_t)H_ * H_);
  cvt_bf16_kernel<<<blk((size_t)R_ * H_), TB, 0, stream>>>(qa, qab, (size_t)R_ * H_);
  cvt_bf16_kernel<<<blk((size_t)R_ * H_), TB, 0, stream>>>(ka, kab, (size_t)R_ * H_);
  cvt_pad_kernel<<<blk((size_t)H_ * RP_), TB, 0, stream>>>(qb, qbp, H_, R_, RP_);
  cvt_pad_kernel<<<blk((size_t)H_ * RP_), TB, 0, stream>>>(kbm, kbp, H_, R_, RP_);
  cvt_bf16_kernel<<<blk((size_t)E_N * H_ * H_), TB, 0, stream>>>(vbase, vbaseb, (size_t)E_N * H_ * H_);
  cvt_bf16_kernel<<<blk((size_t)E_N * H_ * H_), TB, 0, stream>>>(obase, obaseb, (size_t)E_N * H_ * H_);
  cvt_bf16_kernel<<<blk((size_t)E_N * R_ * H_), TB, 0, stream>>>(va, vab, (size_t)E_N * R_ * H_);
  cvt_bf16_kernel<<<blk((size_t)E_N * R_ * H_), TB, 0, stream>>>(oa, oab, (size_t)E_N * R_ * H_);
  cvt_pad_kernel<<<blk((size_t)E_N * H_ * RP_), TB, 0, stream>>>(vb, vbp, E_N * H_, R_, RP_);
  cvt_pad_kernel<<<blk((size_t)E_N * H_ * RP_), TB, 0, stream>>>(ob, obp, E_N * H_, R_, RP_);

  // ---- routing (both gates use hidden_states) ----
  (void)hipMemsetAsync(counts, 0, 2 * E_N * sizeof(int), stream);
  route_kernel<<<blk(M_), TB, 0, stream>>>(x, gv, cntV, btokV, bwV);
  route_kernel<<<blk(M_), TB, 0, stream>>>(x, go, cntO, btokO, bwO);

  // ---- q / k projections (dense, base + LoRA, split-head bf16 output) ----
  dim3 gDense(H_ / 128, M_ / 64, 1);
  gemm_lora_kernel<<<gDense, TB, 0, stream>>>(xb, wqb, qab, qbp, nullptr, qhb,
                                              nullptr, nullptr, nullptr,
                                              M_, H_, H_, SCALING_);
  gemm_lora_kernel<<<gDense, TB, 0, stream>>>(xb, wkb, kab, kbp, nullptr, khb,
                                              nullptr, nullptr, nullptr,
                                              M_, H_, H_, SCALING_);

  // ---- v MoE (grouped top-2, weighted scatter-accumulate) ----
  (void)hipMemsetAsync(vout, 0, (size_t)M_ * H_ * sizeof(float), stream);
  dim3 gMoe(H_ / 128, M_ / 64, E_N);
  gemm_lora_kernel<<<gMoe, TB, 0, stream>>>(xb, vbaseb, vab, vbp, vout, nullptr,
                                            btokV, cntV, bwV, M_, H_, H_, SCALING_);
  vtrans_kernel<<<blk((size_t)M_ * H_), TB, 0, stream>>>(vout, vtb);

  // ---- attention ----
  dim3 gAttn(S_ / 128, NH_, B_);
  attn_kernel<<<gAttn, TB, 0, stream>>>(qhb, khb, vtb, amask, ctxb);

  // ---- o MoE (grouped top-2 on ctx, gates from hidden_states) ----
  (void)hipMemsetAsync(out, 0, (size_t)out_size * sizeof(float), stream);
  gemm_lora_kernel<<<gMoe, TB, 0, stream>>>(ctxb, obaseb, oab, obp, out, nullptr,
                                            btokO, cntO, bwO, M_, H_, H_, SCALING_);

  (void)in_sizes; (void)n_in; (void)ws_size;
}
